// EBTPuzzle_48773648614149
// MI455X (gfx1250) — matrix operand order repaired and checked by reading, not compiled, not run on hardware
//
#include <hip/hip_runtime.h>

// ---------------- problem constants ----------------
#define BB      8
#define SS_     900
#define VV      11
#define HH      512
#define NHH     8
#define HDD     64
#define INNERR  1536
#define NSTEPS  2
#define EPSF    1e-5f
#define THETAF  10000.0f
#define ESCALEF 22.627416997969522f   // sqrt(512)
#define ASCALE  0.125f                // 1/sqrt(64)

typedef __attribute__((ext_vector_type(16))) __bf16 bf16x16;
typedef __attribute__((ext_vector_type(8)))  float  f32x8;

union FragAB { int4 q[2]; bf16x16 v; };

__device__ __forceinline__ unsigned short f2bf(float f) {
  unsigned u = __float_as_uint(f);
  u += 0x7fffu + ((u >> 16) & 1u);          // round-to-nearest-even
  return (unsigned short)(u >> 16);
}

// =====================================================================
// Tiled bf16-WMMA GEMM with fp32 accumulate.
//   C = alpha * op(A) @ op(B) + beta * C
//   TA=0: A is (M,K) row-major, ld=lda.  TA=1: A_eff[m,k] = A[k*lda+m]
//   TB=0: B is (K,N) row-major, ld=ldb.  TB=1: B_eff[k,n] = B[n*ldb+k]
// Block tile 128 x (32*JN); 8 waves (wave32): 4 in M x 2 in N.
// Wave tile 32 x (16*JN)  -> 2 x JN v_wmma_f32_16x16x32_bf16 accumulators.
// LDS double-buffered; fp32->bf16 conversion during staging.
// =====================================================================
template<int TA, int TB, int JN>
__global__ __launch_bounds__(256)
void gemm_wmma(const float* __restrict__ A, const float* __restrict__ B,
               float* __restrict__ C,
               int M, int N, int K, int lda, int ldb, int ldc,
               long long sA, long long sB, long long sC,
               float alpha, float beta)
{
  constexpr int BN = 32 * JN;                 // block N tile (64 or 128)
  constexpr int NQ = BN / 4;                  // n-quads per k column
  __shared__ __align__(16) unsigned short As[2][128 * 40];
  __shared__ __align__(16) unsigned short Bs[2][BN * 40];

  const int z = blockIdx.z;
  A += (long long)z * sA;
  B += (long long)z * sB;
  C += (long long)z * sC;

  const int m0   = blockIdx.x * 128;
  const int n0   = blockIdx.y * BN;
  const int tid  = threadIdx.x;
  const int wave = tid >> 5;
  const int lane = tid & 31;
  const int wm   = wave >> 1;      // 0..3 -> M sub-tile
  const int wn   = wave & 1;       // 0..1 -> N sub-tile
  const int half = lane >> 4;
  const int l16  = lane & 15;
  const bool fa  = (m0 + 128 <= M);
  const bool fb  = (n0 + BN  <= N);

  // ---- stage A tile (fp32 -> bf16) into dst ----
  auto stageA = [&](unsigned short* dst, int k0) {
    if (fa && (k0 + 32 <= K)) {
      if (TA) {
        for (int t = tid; t < 1024; t += 256) {       // float4 along m, scatter k-col
          int c = t >> 5, mq = (t & 31) << 2;
          const float4 v = *reinterpret_cast<const float4*>(
              &A[(long long)(k0 + c) * lda + (m0 + mq)]);
          dst[(mq + 0) * 40 + c] = f2bf(v.x);
          dst[(mq + 1) * 40 + c] = f2bf(v.y);
          dst[(mq + 2) * 40 + c] = f2bf(v.z);
          dst[(mq + 3) * 40 + c] = f2bf(v.w);
        }
        __builtin_prefetch((const void*)&A[(long long)(k0 + 32 + (tid & 31)) * lda + m0], 0, 3);
      } else {
        for (int t = tid; t < 1024; t += 256) {       // float4 along k, packed b64 store
          int r = t >> 3, cq = (t & 7) << 2;
          const float4 v = *reinterpret_cast<const float4*>(
              &A[(long long)(m0 + r) * lda + (k0 + cq)]);
          uint2 pk;
          pk.x = (unsigned)f2bf(v.x) | ((unsigned)f2bf(v.y) << 16);
          pk.y = (unsigned)f2bf(v.z) | ((unsigned)f2bf(v.w) << 16);
          *reinterpret_cast<uint2*>(&dst[r * 40 + cq]) = pk;
        }
        __builtin_prefetch((const void*)&A[(long long)(m0 + (tid & 127)) * lda + (k0 + 32)], 0, 3);
      }
    } else {                                          // edge tiles: branchless clamped loads
      for (int t = tid; t < 4096; t += 256) {
        int r, c;
        if (TA) { r = t & 127; c = t >> 7; }
        else    { r = t >> 5;  c = t & 31; }
        int gm = m0 + r, gk = k0 + c;
        int cm = gm < M ? gm : M - 1;
        int ck = gk < K ? gk : K - 1;
        float v = TA ? A[(long long)ck * lda + cm] : A[(long long)cm * lda + ck];
        v = (gm < M && gk < K) ? v : 0.f;
        dst[r * 40 + c] = f2bf(v);
      }
    }
  };

  // ---- stage B tile K-transposed ([n][k]) into dst ----
  auto stageB = [&](unsigned short* dst, int k0) {
    if (fb && (k0 + 32 <= K)) {
      if (TB) {
        for (int t = tid; t < BN * 8; t += 256) {     // float4 along k, packed b64 store
          int r = t >> 3, cq = (t & 7) << 2;
          const float4 v = *reinterpret_cast<const float4*>(
              &B[(long long)(n0 + r) * ldb + (k0 + cq)]);
          uint2 pk;
          pk.x = (unsigned)f2bf(v.x) | ((unsigned)f2bf(v.y) << 16);
          pk.y = (unsigned)f2bf(v.z) | ((unsigned)f2bf(v.w) << 16);
          *reinterpret_cast<uint2*>(&dst[r * 40 + cq]) = pk;
        }
        __builtin_prefetch((const void*)&B[(long long)(n0 + (tid & (BN - 1))) * ldb + (k0 + 32)], 0, 3);
      } else {
        for (int t = tid; t < BN * 8; t += 256) {     // float4 along n, scatter k-col
          int c = t / NQ, nq = (t % NQ) << 2;
          const float4 v = *reinterpret_cast<const float4*>(
              &B[(long long)(k0 + c) * ldb + (n0 + nq)]);
          dst[(nq + 0) * 40 + c] = f2bf(v.x);
          dst[(nq + 1) * 40 + c] = f2bf(v.y);
          dst[(nq + 2) * 40 + c] = f2bf(v.z);
          dst[(nq + 3) * 40 + c] = f2bf(v.w);
        }
        __builtin_prefetch((const void*)&B[(long long)(k0 + 32 + (tid & 31)) * ldb + n0], 0, 3);
      }
    } else {
      for (int t = tid; t < BN * 32; t += 256) {
        int n, c;
        if (TB) { n = t >> 5;  c = t & 31; }
        else    { n = t % BN;  c = t / BN; }
        int gn = n0 + n, gk = k0 + c;
        int cn = gn < N ? gn : N - 1;
        int ck = gk < K ? gk : K - 1;
        float v = TB ? B[(long long)cn * ldb + ck] : B[(long long)ck * ldb + cn];
        v = (gn < N && gk < K) ? v : 0.f;
        dst[n * 40 + c] = f2bf(v);
      }
    }
  };

  f32x8 acc[2][JN];
  for (int i = 0; i < 2; i++)
    for (int j = 0; j < JN; j++)
      for (int e = 0; e < 8; e++) acc[i][j][e] = 0.f;

  const int nk = (K + 31) >> 5;
  stageA(As[0], 0);
  stageB(Bs[0], 0);

  for (int kt = 0; kt < nk; ++kt) {
    __syncthreads();                        // staging for tile kt complete
    const int cur = kt & 1;
    if (kt + 1 < nk) {                      // overlap: stage kt+1 into other buffer
      stageA(As[cur ^ 1], (kt + 1) * 32);
      stageB(Bs[cur ^ 1], (kt + 1) * 32);
    }
    const unsigned short* Ac = As[cur];
    const unsigned short* Bc = Bs[cur];

    FragAB af[2], bfrag[JN];
    #pragma unroll
    for (int i = 0; i < 2; i++) {
      int row = wm * 32 + i * 16 + l16;
      af[i].q[0] = *reinterpret_cast<const int4*>(&Ac[row * 40 + 8 * half]);
      af[i].q[1] = *reinterpret_cast<const int4*>(&Ac[row * 40 + 16 + 8 * half]);
    }
    #pragma unroll
    for (int j = 0; j < JN; j++) {
      int col = wn * (16 * JN) + j * 16 + l16;
      const int4* pb = reinterpret_cast<const int4*>(&Bc[col * 40 + 8 * half]);
      bfrag[j].q[0] = pb[0];
      bfrag[j].q[1] = pb[2];
    }
    #pragma unroll
    for (int i = 0; i < 2; i++)
      #pragma unroll
      for (int j = 0; j < JN; j++)
        acc[i][j] = __builtin_amdgcn_wmma_f32_16x16x32_bf16(
            false, af[i].v, false, bfrag[j].v, (short)0, acc[i][j], false, false);
  }

  // ---- epilogue: D layout VGPR v -> M = v + 8*half, N = l16 ----
  if (fa && fb) {
    for (int i = 0; i < 2; i++) {
      int rbase = m0 + wm * 32 + i * 16 + half * 8;
      for (int j = 0; j < JN; j++) {
        int colg = n0 + wn * (16 * JN) + j * 16 + l16;
        for (int v = 0; v < 8; v++) {
          long long off = (long long)(rbase + v) * ldc + colg;
          float r = alpha * acc[i][j][v];
          if (beta != 0.f) r += beta * C[off];
          C[off] = r;
        }
      }
    }
  } else {
    for (int i = 0; i < 2; i++) {
      int rbase = m0 + wm * 32 + i * 16 + half * 8;
      for (int j = 0; j < JN; j++) {
        int colg = n0 + wn * (16 * JN) + j * 16 + l16;
        if (colg < N) {
          for (int v = 0; v < 8; v++) {
            int rowg = rbase + v;
            if (rowg < M) {
              long long off = (long long)rowg * ldc + colg;
              float r = alpha * acc[i][j][v];
              if (beta != 0.f) r += beta * C[off];
              C[off] = r;
            }
          }
        }
      }
    }
  }
}

// =====================================================================
// Elementwise / reduction kernels (fp32)
// =====================================================================
__global__ void k_zero(float* p, long long n) {
  long long i = (long long)blockIdx.x * 256 + threadIdx.x;
  if (i < n) p[i] = 0.f;
}
__global__ void k_copy(float* dst, const float* src, long long n) {
  long long i = (long long)blockIdx.x * 256 + threadIdx.x;
  if (i < n) dst[i] = src[i];
}
__global__ void k_rope_tables(float* cosb, float* sinb) {
  int idx = blockIdx.x * 256 + threadIdx.x;
  if (idx >= SS_ * HDD) return;
  int s = idx / HDD, d = idx % HDD;
  int i = d & 31;                                   // concat(freqs,freqs)
  float invf = powf(THETAF, -(float)(2 * i) / (float)HDD);
  float ang = (float)s * invf;
  cosb[idx] = cosf(ang);
  sinb[idx] = sinf(ang);
}
__global__ void k_embed(const int* tok, const float* ew, float* out) {
  long long idx = (long long)blockIdx.x * 256 + threadIdx.x;
  if (idx >= (long long)BB * SS_ * HH) return;
  long long row = idx / HH;
  int j = (int)(idx % HH);
  out[idx] = ESCALEF * ew[(long long)tok[row] * HH + j];
}
__global__ void k_softmax_small(const float* logits, float* probs) {
  int r = blockIdx.x * 256 + threadIdx.x;
  if (r >= BB * SS_) return;
  const float* x = logits + (long long)r * VV;
  float mx = x[0];
  for (int i = 1; i < VV; i++) mx = fmaxf(mx, x[i]);
  float e[VV], s = 0.f;
  for (int i = 0; i < VV; i++) { e[i] = __expf(x[i] - mx); s += e[i]; }
  float inv = 1.f / s;
  float* p = probs + (long long)r * VV;
  for (int i = 0; i < VV; i++) p[i] = e[i] * inv;
}
__global__ void k_rope_fwd(float* qkv, const float* cosb, const float* sinb) {
  int idx = blockIdx.x * 256 + threadIdx.x;
  if (idx >= BB * SS_ * NHH * 32) return;
  int p = idx & 31;
  int h = (idx >> 5) & 7;
  int s = (idx >> 8) % SS_;
  int b = (idx >> 8) / SS_;
  long long base = ((long long)(b * SS_ + s)) * 1536 + h * 64;
  float c0 = cosb[s * 64 + p], c1 = cosb[s * 64 + p + 32];
  float s0 = sinb[s * 64 + p], s1 = sinb[s * 64 + p + 32];
  float q0 = qkv[base + p], q1 = qkv[base + p + 32];
  qkv[base + p]      = q0 * c0 - q1 * s0;
  qkv[base + p + 32] = q1 * c1 + q0 * s1;
  float k0 = qkv[base + 512 + p], k1 = qkv[base + 512 + p + 32];
  qkv[base + 512 + p]      = k0 * c0 - k1 * s0;
  qkv[base + 512 + p + 32] = k1 * c1 + k0 * s1;
}
__global__ void k_rope_bwd(float* dqkv, const float* cosb, const float* sinb) {
  int idx = blockIdx.x * 256 + threadIdx.x;
  if (idx >= BB * SS_ * NHH * 32) return;
  int p = idx & 31;
  int h = (idx >> 5) & 7;
  int s = (idx >> 8) % SS_;
  int b = (idx >> 8) / SS_;
  long long base = ((long long)(b * SS_ + s)) * 1536 + h * 64;
  float c0 = cosb[s * 64 + p], c1 = cosb[s * 64 + p + 32];
  float s0 = sinb[s * 64 + p], s1 = sinb[s * 64 + p + 32];
  float q0 = dqkv[base + p], q1 = dqkv[base + p + 32];
  dqkv[base + p]      = q0 * c0 + q1 * s1;    // R^T = -R
  dqkv[base + p + 32] = q1 * c1 - q0 * s0;
  float k0 = dqkv[base + 512 + p], k1 = dqkv[base + 512 + p + 32];
  dqkv[base + 512 + p]      = k0 * c0 + k1 * s1;
  dqkv[base + 512 + p + 32] = k1 * c1 - k0 * s0;
}
__global__ void k_softmax_rows(float* buf, int n, int ld) {
  __shared__ float red[256];
  float* p = buf + (long long)blockIdx.x * ld;
  int t = threadIdx.x;
  float mx = -3.4e38f;
  for (int i = t; i < n; i += 256) mx = fmaxf(mx, p[i]);
  red[t] = mx; __syncthreads();
  for (int s = 128; s > 0; s >>= 1) { if (t < s) red[t] = fmaxf(red[t], red[t + s]); __syncthreads(); }
  mx = red[0]; __syncthreads();
  float sum = 0.f;
  for (int i = t; i < n; i += 256) { float e = __expf(p[i] - mx); p[i] = e; sum += e; }
  red[t] = sum; __syncthreads();
  for (int s = 128; s > 0; s >>= 1) { if (t < s) red[t] += red[t + s]; __syncthreads(); }
  float inv = 1.f / red[0];
  for (int i = t; i < n; i += 256) p[i] *= inv;
}
__global__ void k_softmax_bwd_rows(float* d, const float* a, int n, int ld) {
  __shared__ float red[256];
  long long off = (long long)blockIdx.x * ld;
  float* dp = d + off;
  const float* ap = a + off;
  int t = threadIdx.x;
  float dot = 0.f;
  for (int i = t; i < n; i += 256) dot += dp[i] * ap[i];
  red[t] = dot; __syncthreads();
  for (int s = 128; s > 0; s >>= 1) { if (t < s) red[t] += red[t + s]; __syncthreads(); }
  dot = red[0];
  for (int i = t; i < n; i += 256) dp[i] = ap[i] * (dp[i] - dot);
}
// y = (a+b) * rsqrt(mean((a+b)^2)+eps); store y and inv per row (H=512)
__global__ void k_rmsnorm_fwd(const float* a, const float* b, float* y, float* invb) {
  __shared__ float red[256];
  int row = blockIdx.x, t = threadIdx.x;
  const float* ar = a + (long long)row * 512;
  const float* br = b + (long long)row * 512;
  float x0 = ar[t] + br[t];
  float x1 = ar[t + 256] + br[t + 256];
  red[t] = x0 * x0 + x1 * x1; __syncthreads();
  for (int s = 128; s > 0; s >>= 1) { if (t < s) red[t] += red[t + s]; __syncthreads(); }
  float iv = rsqrtf(red[0] / 512.f + EPSF);
  float* yr = y + (long long)row * 512;
  yr[t] = x0 * iv;
  yr[t + 256] = x1 * iv;
  if (t == 0) invb[row] = iv;
}
// dx = inv * (dy - y * mean(dy*y))
__global__ void k_rmsnorm_bwd(const float* dy, const float* y, const float* invb, float* dx) {
  __shared__ float red[256];
  int row = blockIdx.x, t = threadIdx.x;
  const float* dyr = dy + (long long)row * 512;
  const float* yr  = y  + (long long)row * 512;
  red[t] = dyr[t] * yr[t] + dyr[t + 256] * yr[t + 256]; __syncthreads();
  for (int s = 128; s > 0; s >>= 1) { if (t < s) red[t] += red[t + s]; __syncthreads(); }
  float dot = red[0] / 512.f;
  float iv = invb[row];
  float* dxr = dx + (long long)row * 512;
  dxr[t]       = iv * (dyr[t]       - yr[t]       * dot);
  dxr[t + 256] = iv * (dyr[t + 256] - yr[t + 256] * dot);
}
__global__ void k_silu_fwd(const float* gu, float* act) {
  long long idx = (long long)blockIdx.x * 256 + threadIdx.x;
  if (idx >= (long long)BB * SS_ * INNERR) return;
  long long row = idx / INNERR;
  int i = (int)(idx % INNERR);
  float g = gu[row * 3072 + i];
  float u = gu[row * 3072 + 1536 + i];
  float sg = 1.f / (1.f + __expf(-g));
  act[idx] = g * sg * u;
}
__global__ void k_silu_bwd(const float* dact, const float* gu, float* dgu) {
  long long idx = (long long)blockIdx.x * 256 + threadIdx.x;
  if (idx >= (long long)BB * SS_ * INNERR) return;
  long long row = idx / INNERR;
  int i = (int)(idx % INNERR);
  float da = dact[idx];
  float g = gu[row * 3072 + i];
  float u = gu[row * 3072 + 1536 + i];
  float sg = 1.f / (1.f + __expf(-g));
  dgu[row * 3072 + i]        = da * u * sg * (1.f + g * (1.f - sg));
  dgu[row * 3072 + 1536 + i] = da * g * sg;
}
__global__ void k_energy_grad(float* dh, const float* ew) {
  long long idx = (long long)blockIdx.x * 256 + threadIdx.x;
  if (idx >= (long long)BB * SS_ * HH) return;
  dh[idx] = ew[(int)(idx & 511)];
}
__global__ void k_logits_update(float* logits, const float* probs,
                                const float* dprobs, const float* alpha) {
  int r = blockIdx.x * 256 + threadIdx.x;
  if (r >= BB * SS_) return;
  float a = fmaxf(alpha[0], 1e-4f);
  const float* p  = probs  + (long long)r * VV;
  const float* dp = dprobs + (long long)r * VV;
  float dot = 0.f;
  for (int i = 0; i < VV; i++) dot += p[i] * dp[i];
  float* lg = logits + (long long)r * VV;
  for (int i = 0; i < VV; i++) lg[i] -= a * p[i] * (dp[i] - dot);
}

// =====================================================================
// Host-side orchestration
// =====================================================================
static void gemm(hipStream_t st, int TA, int TB,
                 const float* A, const float* B, float* C,
                 int M, int N, int K, int lda, int ldb, int ldc,
                 long long sA, long long sB, long long sC, int batch,
                 float alpha, float beta)
{
  dim3 blk(256);
  if (N > 64) {
    dim3 g((M + 127) / 128, (N + 127) / 128, batch);
    if (!TA && !TB)
      gemm_wmma<0, 0, 4><<<g, blk, 0, st>>>(A, B, C, M, N, K, lda, ldb, ldc, sA, sB, sC, alpha, beta);
    else if (!TA && TB)
      gemm_wmma<0, 1, 4><<<g, blk, 0, st>>>(A, B, C, M, N, K, lda, ldb, ldc, sA, sB, sC, alpha, beta);
    else
      gemm_wmma<1, 0, 4><<<g, blk, 0, st>>>(A, B, C, M, N, K, lda, ldb, ldc, sA, sB, sC, alpha, beta);
  } else {
    dim3 g((M + 127) / 128, (N + 63) / 64, batch);
    if (!TA && !TB)
      gemm_wmma<0, 0, 2><<<g, blk, 0, st>>>(A, B, C, M, N, K, lda, ldb, ldc, sA, sB, sC, alpha, beta);
    else if (!TA && TB)
      gemm_wmma<0, 1, 2><<<g, blk, 0, st>>>(A, B, C, M, N, K, lda, ldb, ldc, sA, sB, sC, alpha, beta);
    else
      gemm_wmma<1, 0, 2><<<g, blk, 0, st>>>(A, B, C, M, N, K, lda, ldb, ldc, sA, sB, sC, alpha, beta);
  }
}

extern "C" void kernel_launch(void* const* d_in, const int* in_sizes, int n_in,
                              void* d_out, int out_size, void* d_ws, size_t ws_size,
                              hipStream_t stream)
{
  (void)in_sizes; (void)n_in; (void)out_size; (void)ws_size;
  const int*   tokens   = (const int*)d_in[0];
  // d_in[1] = puzzle_identifiers (unused by the reference math)
  const float* embed_w  = (const float*)d_in[2];   // (V,H)
  const float* energy_w = (const float*)d_in[3];   // (H,1)
  const float* qkv_w    = (const float*)d_in[4];   // (L,H,1536)
  const float* o_w      = (const float*)d_in[5];   // (L,512,512)
  const float* gu_w     = (const float*)d_in[6];   // (L,512,3072)
  const float* d_w      = (const float*)d_in[7];   // (L,1536,512)
  const float* alpha    = (const float*)d_in[8];
  float* logits = (float*)d_out;                   // (B,S,V) fp32

  const long long BS = (long long)BB * SS_;        // 7200

  float* ws = (float*)d_ws;
  auto take = [&](long long n) { float* p = ws; ws += n; return p; };

  float* cosb      = take((long long)SS_ * HDD);
  float* sinb      = take((long long)SS_ * HDD);
  float* input_emb = take(BS * HH);
  float* h0        = take(BS * HH);
  float* qkvbuf[2] = { take(BS * 1536), take(BS * 1536) };   // roped q,k + v (saved)
  float* h1buf[2]  = { take(BS * HH),   take(BS * HH) };
  float* houtbuf[2]= { take(BS * HH),   take(BS * HH) };
  float* gubuf[2]  = { take(BS * 3072), take(BS * 3072) };
  float* inv1b[2]  = { take(BS), take(BS) };
  float* inv2b[2]  = { take(BS), take(BS) };
  float* probs     = take(BS * VV);
  float* scoresb   = take((long long)NHH * SS_ * SS_);   // per-b attn scores
  float* dscoresb  = take((long long)NHH * SS_ * SS_);
  float* dh        = take(BS * HH);
  float* dtmp      = take(BS * HH);
  float* doutb     = take(BS * HH);
  float* dqkv      = take(BS * 1536);
  float* dgu       = take(BS * 3072);
  float* dact      = take(BS * 1536);
  float* dprobs    = take(BS * VV);
  // forward-only temps alias backward buffers (disjoint lifetimes):
  float* outb  = doutb;   // attention output (B,S,512)
  float* o_tmp = dtmp;    // out @ o_w
  float* act   = dact;    // silu(gate)*up

  auto ew = [](long long n) { return dim3((unsigned)((n + 255) / 256)); };
  const long long SxS = (long long)SS_ * SS_;   // 810000

  k_rope_tables<<<ew((long long)SS_ * HDD), 256, 0, stream>>>(cosb, sinb);
  k_embed<<<ew(BS * HH), 256, 0, stream>>>(tokens, embed_w, input_emb);
  k_zero<<<ew(BS * VV), 256, 0, stream>>>(logits, BS * VV);   // logits start at 0

  for (int step = 0; step < NSTEPS; ++step) {
    // ---------------- forward ----------------
    k_softmax_small<<<ew(BS), 256, 0, stream>>>(logits, probs);
    k_copy<<<ew(BS * HH), 256, 0, stream>>>(h0, input_emb, BS * HH);
    gemm(stream, 0, 0, probs, embed_w, h0, 7200, 512, 11, 11, 512, 512,
         0, 0, 0, 1, ESCALEF, 1.f);                       // h0 += ESCALE*probs@E

    for (int l = 0; l < 2; l++) {
      const float* hin = l ? houtbuf[0] : h0;
      float* qkvl = qkvbuf[l];
      gemm(stream, 0, 0, hin, qkv_w + (long long)l * HH * 1536, qkvl,
           7200, 1536, 512, 512, 1536, 1536, 0, 0, 0, 1, 1.f, 0.f);
      k_rope_fwd<<<ew(BS * NHH * 32), 256, 0, stream>>>(qkvl, cosb, sinb);
      for (int b = 0; b < BB; b++) {
        long long qb = (long long)b * SS_ * 1536;
        gemm(stream, 0, 1, qkvl + qb, qkvl + qb + 512, scoresb,       // S = scale*Q@K^T
             900, 900, 64, 1536, 1536, 900, 64, 64, SxS, 8, ASCALE, 0.f);
        k_softmax_rows<<<dim3(NHH * SS_), 256, 0, stream>>>(scoresb, 900, 900);
        gemm(stream, 0, 0, scoresb, qkvl + qb + 1024,                 // out = P@V
             outb + (long long)b * SS_ * HH,
             900, 64, 900, 900, 1536, 512, SxS, 64, 64, 8, 1.f, 0.f);
      }
      gemm(stream, 0, 0, outb, o_w + (long long)l * HH * HH, o_tmp,
           7200, 512, 512, 512, 512, 512, 0, 0, 0, 1, 1.f, 0.f);
      k_rmsnorm_fwd<<<dim3(7200), 256, 0, stream>>>(hin, o_tmp, h1buf[l], inv1b[l]);
      gemm(stream, 0, 0, h1buf[l], gu_w + (long long)l * HH * 3072, gubuf[l],
           7200, 3072, 512, 512, 3072, 3072, 0, 0, 0, 1, 1.f, 0.f);
      k_silu_fwd<<<ew(BS * INNERR), 256, 0, stream>>>(gubuf[l], act);
      gemm(stream, 0, 0, act, d_w + (long long)l * INNERR * HH, dh,   // dh = mlp temp
           7200, 512, 1536, 1536, 512, 512, 0, 0, 0, 1, 1.f, 0.f);
      k_rmsnorm_fwd<<<dim3(7200), 256, 0, stream>>>(h1buf[l], dh, houtbuf[l], inv2b[l]);
    }

    // ---------------- backward (d energy / d logits) ----------------
    k_energy_grad<<<ew(BS * HH), 256, 0, stream>>>(dh, energy_w);     // dE/dh_final
    for (int l = 1; l >= 0; l--) {
      float* qkvl = qkvbuf[l];
      k_rmsnorm_bwd<<<dim3(7200), 256, 0, stream>>>(dh, houtbuf[l], inv2b[l], dtmp); // dr2
      gemm(stream, 0, 1, dtmp, d_w + (long long)l * INNERR * HH, dact,
           7200, 1536, 512, 512, 512, 1536, 0, 0, 0, 1, 1.f, 0.f);    // dact = dr2@dW^T
      k_silu_bwd<<<ew(BS * INNERR), 256, 0, stream>>>(dact, gubuf[l], dgu);
      k_copy<<<ew(BS * HH), 256, 0, stream>>>(dh, dtmp, BS * HH);
      gemm(stream, 0, 1, dgu, gu_w + (long long)l * HH * 3072, dh,
           7200, 512, 3072, 3072, 3072, 512, 0, 0, 0, 1, 1.f, 1.f);   // dh1
      k_rmsnorm_bwd<<<dim3(7200), 256, 0, stream>>>(dh, h1buf[l], inv1b[l], dtmp);  // dr1
      gemm(stream, 0, 1, dtmp, o_w + (long long)l * HH * HH, doutb,
           7200, 512, 512, 512, 512, 512, 0, 0, 0, 1, 1.f, 0.f);      // dout
      for (int b = 0; b < BB; b++) {
        long long qb = (long long)b * SS_ * 1536;
        long long ob = (long long)b * SS_ * HH;
        gemm(stream, 0, 1, qkvl + qb, qkvl + qb + 512, scoresb,       // recompute P
             900, 900, 64, 1536, 1536, 900, 64, 64, SxS, 8, ASCALE, 0.f);
        k_softmax_rows<<<dim3(NHH * SS_), 256, 0, stream>>>(scoresb, 900, 900);
        gemm(stream, 0, 1, doutb + ob, qkvl + qb + 1024, dscoresb,    // dP = dout@V^T
             900, 900, 64, 512, 1536, 900, 64, 64, SxS, 8, 1.f, 0.f);
        k_softmax_bwd_rows<<<dim3(NHH * SS_), 256, 0, stream>>>(dscoresb, scoresb, 900, 900);
        gemm(stream, 1, 0, scoresb, doutb + ob, dqkv + qb + 1024,     // dV = P^T@dout
             900, 64, 900, 900, 512, 1536, SxS, 64, 64, 8, 1.f, 0.f);
        gemm(stream, 0, 0, dscoresb, qkvl + qb + 512, dqkv + qb,      // dQr = s*dS@Kr
             900, 64, 900, 900, 1536, 1536, SxS, 64, 64, 8, ASCALE, 0.f);
        gemm(stream, 1, 0, dscoresb, qkvl + qb, dqkv + qb + 512,      // dKr = s*dS^T@Qr
             900, 64, 900, 900, 1536, 1536, SxS, 64, 64, 8, ASCALE, 0.f);
      }
      k_rope_bwd<<<ew(BS * NHH * 32), 256, 0, stream>>>(dqkv, cosb, sinb);
      k_copy<<<ew(BS * HH), 256, 0, stream>>>(dh, dtmp, BS * HH);
      gemm(stream, 0, 1, dqkv, qkv_w + (long long)l * HH * 1536, dh,
           7200, 512, 1536, 1536, 1536, 512, 0, 0, 0, 1, 1.f, 1.f);   // dh_in
    }
    gemm(stream, 0, 1, dh, embed_w, dprobs,                           // ESCALE*dh0@E^T
         7200, 11, 512, 512, 512, 11, 0, 0, 0, 1, ESCALEF, 0.f);
    k_logits_update<<<ew(BS), 256, 0, stream>>>(logits, probs, dprobs, alpha);
  }
}
